// TransformerBlock_60181081751894
// MI455X (gfx1250) — compile-verified, matrix operand order checked by reading, not run on hardware
//
#include <hip/hip_runtime.h>
#include <hip/hip_bf16.h>

// ---------------------------------------------------------------------------
// Types for CDNA5 WMMA (wave32): V_WMMA_F32_16X16X32_BF16
// ---------------------------------------------------------------------------
typedef __bf16 bf16;
typedef __attribute__((ext_vector_type(16))) __bf16 v16bf;
typedef __attribute__((ext_vector_type(8)))  __bf16 v8bf;
typedef __attribute__((ext_vector_type(8)))  float  v8f;

union Frag16 {            // one WMMA 16-bit operand fragment = two b128 loads
  v16bf v;
  v8bf  h[2];
};

__device__ __forceinline__ v8f wmma_bf16f32(v16bf a, v16bf b, v8f c) {
  // 8 args: (neg_a, A, neg_b, B, c_mod, C, reuse_a, reuse_b)
  return __builtin_amdgcn_wmma_f32_16x16x32_bf16(
      false, a, false, b, (short)0, c, false, false);
}

// Problem dims (fixed by the reference)
#define BB   2
#define SS   2048
#define DD   1024
#define HH   16
#define HDD  64
#define DFF_ 4096
#define MM   (BB * SS)          // 4096 token rows

// ---------------------------------------------------------------------------
// fp32 -> bf16 conversion (grid-stride), plain row-major
// ---------------------------------------------------------------------------
__global__ void cvt_f32_bf16(const float* __restrict__ in, bf16* __restrict__ out,
                             size_t n) {
  for (size_t i = (size_t)blockIdx.x * blockDim.x + threadIdx.x; i < n;
       i += (size_t)gridDim.x * blockDim.x)
    out[i] = (bf16)in[i];
}

// ---------------------------------------------------------------------------
// fp32 -> bf16 weight conversion INTO WMMA B-fragment-major order:
//   Wsw[(ktile*(N/16) + ntile)*512 + lane*16 + e]
// where k = ktile*32 + kh + e + (e>=8?8:0), kh = (lane>=16)?8:0,
//       n = ntile*16 + (lane&15).
// Each lane's 16 fragment elements are contiguous (32B, 16B-aligned) so the
// GEMM loads a B fragment as two coalesced global_load_b128.
// ---------------------------------------------------------------------------
__global__ void cvt_swizzle_w(const float* __restrict__ W, bf16* __restrict__ Wsw,
                              int K, int N) {
  const size_t total = (size_t)K * N;
  const int ntiles = N >> 4;
  for (size_t i = (size_t)blockIdx.x * blockDim.x + threadIdx.x; i < total;
       i += (size_t)gridDim.x * blockDim.x) {
    int e     = (int)(i & 15);
    int lane  = (int)((i >> 4) & 31);
    size_t ch = i >> 9;
    int nt    = (int)(ch % ntiles);
    int ktile = (int)(ch / ntiles);
    int kh    = (lane >> 4) ? 8 : 0;
    int k     = ktile * 32 + kh + e + (e >= 8 ? 8 : 0);
    int n     = nt * 16 + (lane & 15);
    Wsw[i] = (bf16)W[(size_t)k * N + n];
  }
}

// ---------------------------------------------------------------------------
// LDS-free bf16 WMMA GEMM: C[M,N] = A[M,K] @ W[K,N] + bias (+ReLU), output
// f32 or bf16. A row-major bf16; W fragment-swizzled (cvt_swizzle_w).
// Block = 128 threads = 4 waves; block tile 128x64, wave computes 32x64
// (2 A-fragments x 4 B-fragments = 8 WMMA tiles). Per K-step: 12 b128 loads
// issued up-front, then 8 WMMAs -> partial loadcnt waits, good overlap.
// ---------------------------------------------------------------------------
__global__ void gemm_bf16(const bf16* __restrict__ A, const bf16* __restrict__ Wsw,
                          const float* __restrict__ bias,
                          float* __restrict__ Cf, bf16* __restrict__ Cb,
                          int M, int N, int K, int relu) {
  const int tid  = threadIdx.x;          // 0..127
  const int wave = tid >> 5;             // 0..3
  const int lane = tid & 31;
  const int ln15 = lane & 15;
  const int half = lane >> 4;
  const int kh   = half ? 8 : 0;
  const int m0   = blockIdx.y * 128;
  const int n0   = blockIdx.x * 64;
  const int nt0  = n0 >> 4;
  const int ntiles = N >> 4;

  v8f acc[2][4] = {};

  const bf16* arow0 = A + (size_t)(m0 + 32 * wave + ln15) * K + kh;
  const bf16* arow1 = arow0 + (size_t)16 * K;

  for (int kt = 0; kt < K; kt += 32) {
    // Two A fragments (rows strip + 16) — 4 b128 loads.
    Frag16 af[2];
    af[0].h[0] = *(const v8bf*)(arow0 + kt);
    af[0].h[1] = *(const v8bf*)(arow0 + kt + 16);
    af[1].h[0] = *(const v8bf*)(arow1 + kt);
    af[1].h[1] = *(const v8bf*)(arow1 + kt + 16);

    const bf16* wbase =
        Wsw + (((size_t)(kt >> 5) * ntiles + nt0) << 9) + (lane << 4);
    __builtin_prefetch(arow0 + kt + 32, 0, 1);
    __builtin_prefetch(arow1 + kt + 32, 0, 1);
    __builtin_prefetch(wbase + (size_t)ntiles * 512, 0, 1);

    // Four B fragments — 8 coalesced b128 loads, issued before the WMMA burst.
    Frag16 bfr[4];
#pragma unroll
    for (int j = 0; j < 4; ++j) {
      bfr[j].h[0] = *(const v8bf*)(wbase + j * 512);
      bfr[j].h[1] = *(const v8bf*)(wbase + j * 512 + 8);
    }

#pragma unroll
    for (int j = 0; j < 4; ++j) {
      acc[0][j] = wmma_bf16f32(af[0].v, bfr[j].v, acc[0][j]);
      acc[1][j] = wmma_bf16f32(af[1].v, bfr[j].v, acc[1][j]);
    }
  }

  // Epilogue: C/D tile layout — row = r + 8*half, col = lane&15.
#pragma unroll
  for (int i = 0; i < 2; ++i) {
#pragma unroll
    for (int j = 0; j < 4; ++j) {
#pragma unroll
      for (int r = 0; r < 8; ++r) {
        int row = m0 + 32 * wave + 16 * i + r + 8 * half;
        int col = n0 + 16 * j + ln15;
        float v = acc[i][j][r];
        if (bias) v += bias[col];
        if (relu) v = fmaxf(v, 0.0f);
        if (Cb) Cb[(size_t)row * N + col] = (bf16)v;
        else    Cf[(size_t)row * N + col] = v;
      }
    }
  }
}

// ---------------------------------------------------------------------------
// PoPE: softplus magnitudes + rotary phases -> q_p/k_p (B,H,S,128) bf16, and
// V TRANSPOSED (B,H,HD,S) bf16 so attention V-fragments are contiguous loads.
// ---------------------------------------------------------------------------
__global__ void pope_kernel(const float* __restrict__ q, const float* __restrict__ k,
                            const float* __restrict__ v, const int* __restrict__ pos,
                            const float* __restrict__ pbias,
                            bf16* __restrict__ qp, bf16* __restrict__ kp,
                            bf16* __restrict__ vpT) {
  const float TWO_PI = 6.28318530717958647f;
  const size_t n = (size_t)BB * SS * DD;  // 2^22
  for (size_t i = (size_t)blockIdx.x * blockDim.x + threadIdx.x; i < n;
       i += (size_t)gridDim.x * blockDim.x) {
    int hd = (int)(i & 63);
    int h  = (int)((i >> 6) & 15);
    int s  = (int)((i >> 10) & 2047);
    int b  = (int)(i >> 21);
    int bh = b * HH + h;
    // freq = 10000^(-hd/64) = exp(-hd * ln(1e4)/64)
    float freq = __expf(-0.143911568f * (float)hd);
    float ph   = (float)pos[s] * freq;
    float qv = q[i], kv = k[i];
    float mq = qv > 20.f ? qv : log1pf(__expf(qv));
    float mk = kv > 20.f ? kv : log1pf(__expf(kv));
    float cb = pbias[h * HDD + hd];
    cb = fminf(0.f, fmaxf(-TWO_PI, cb));
    float kph = ph + cb;
    float cq, sq, ck, sk;
    __sincosf(ph, &sq, &cq);
    __sincosf(kph, &sk, &ck);
    size_t o = ((size_t)bh * SS + s) * 128 + hd;
    qp[o]       = (bf16)(mq * cq);
    qp[o + HDD] = (bf16)(mq * sq);
    kp[o]       = (bf16)(mk * ck);
    kp[o + HDD] = (bf16)(mk * sk);
    vpT[((size_t)bh * HDD + hd) * SS + s] = (bf16)v[i];
  }
}

// ---------------------------------------------------------------------------
// Flash-style causal attention with softmax1 (denom += exp(-m)); all matmuls
// on WMMA. One wave per 16-query tile per (b,h). Score K-dim = 128.
// Q/K/V fragments load straight from global as b128 pairs; only the P tile
// goes through LDS (row stride 64B -> fragment reads are 2x ds_load_b128).
// All fragment loads for a burst are issued before the WMMAs.
// ---------------------------------------------------------------------------
__global__ void attention_pope(const bf16* __restrict__ qp_all,
                               const bf16* __restrict__ kp_all,
                               const bf16* __restrict__ vpT_all,
                               bf16* __restrict__ attnb) {
  const float scale = 0.088388347648f;   // 1/sqrt(2*HD)
  const int bh = blockIdx.y;             // 0..B*H-1
  const int b = bh >> 4, h = bh & 15;
  const int qt = blockIdx.x;
  const int qrow0 = qt * 16;
  const int lane = threadIdx.x & 31;
  const int ln15 = lane & 15;
  const int half = lane >> 4;
  const int kh   = half ? 8 : 0;

  const bf16* qp  = qp_all  + (size_t)bh * SS * 128;
  const bf16* kp  = kp_all  + (size_t)bh * SS * 128;
  const bf16* vpT = vpT_all + (size_t)bh * HDD * SS;

  __shared__ bf16 pt[16][32];    // P tile staging (re-layout for A fragment)

  // Load Q fragments once: 16 rows x 128 K -> 4 A-fragments (b128 pairs).
  Frag16 qf[4];
  const bf16* qrow = qp + (size_t)(qrow0 + ln15) * 128 + kh;
#pragma unroll
  for (int t = 0; t < 4; ++t) {
    qf[t].h[0] = *(const v8bf*)(qrow + 32 * t);
    qf[t].h[1] = *(const v8bf*)(qrow + 32 * t + 16);
  }

  float m[8], l[8];
#pragma unroll
  for (int r = 0; r < 8; ++r) { m[r] = -1e30f; l[r] = 0.f; }
  v8f acc[4] = {};

  for (int kb = 0; kb <= qrow0; kb += 32) {
    // Preload all K fragments for both 16-key tiles (16 b128 loads).
    const bf16* k1p = kp + (size_t)(kb + ln15) * 128 + kh;
    const bf16* k2p = kp + (size_t)(kb + 16 + ln15) * 128 + kh;
    __builtin_prefetch(k1p + 32 * 128, 0, 1);
    __builtin_prefetch(k2p + 32 * 128, 0, 1);
    Frag16 kf1[4], kf2[4];
#pragma unroll
    for (int t = 0; t < 4; ++t) {
      kf1[t].h[0] = *(const v8bf*)(k1p + 32 * t);
      kf1[t].h[1] = *(const v8bf*)(k1p + 32 * t + 16);
      kf2[t].h[0] = *(const v8bf*)(k2p + 32 * t);
      kf2[t].h[1] = *(const v8bf*)(k2p + 32 * t + 16);
    }

    v8f s1 = {}, s2 = {};
#pragma unroll
    for (int t = 0; t < 4; ++t) {
      s1 = wmma_bf16f32(qf[t].v, kf1[t].v, s1);
      s2 = wmma_bf16f32(qf[t].v, kf2[t].v, s2);
    }

    // Causal mask + online softmax1 update. Row r+8*half, key col = lane&15.
    float p1[8], p2[8], resc[8];
#pragma unroll
    for (int r = 0; r < 8; ++r) {
      int row = qrow0 + r + 8 * half;
      float v1 = s1[r] * scale;
      float v2 = s2[r] * scale;
      if (kb + ln15 > row)      v1 = -1e30f;
      if (kb + 16 + ln15 > row) v2 = -1e30f;
      float mx = fmaxf(v1, v2);
#pragma unroll
      for (int off = 8; off; off >>= 1) mx = fmaxf(mx, __shfl_xor(mx, off, 32));
      float mn = fmaxf(m[r], mx);
      float rs = __expf(m[r] - mn);
      p1[r] = __expf(v1 - mn);
      p2[r] = __expf(v2 - mn);
      float sm = p1[r] + p2[r];
#pragma unroll
      for (int off = 8; off; off >>= 1) sm += __shfl_xor(sm, off, 32);
      l[r] = l[r] * rs + sm;
      m[r] = mn;
      resc[r] = rs;
    }
#pragma unroll
    for (int j = 0; j < 4; ++j)
#pragma unroll
      for (int r = 0; r < 8; ++r) acc[j][r] *= resc[r];

    // Re-layout P (16 queries x 32 keys) through LDS into A-fragment form.
    __syncthreads();
#pragma unroll
    for (int r = 0; r < 8; ++r) {
      pt[r + 8 * half][ln15]      = (bf16)p1[r];
      pt[r + 8 * half][16 + ln15] = (bf16)p2[r];
    }
    __syncthreads();

    Frag16 pf;
    pf.h[0] = *(const v8bf*)&pt[ln15][kh];
    pf.h[1] = *(const v8bf*)&pt[ln15][kh + 16];

    // P @ V: preload 4 V fragments (8 b128 loads), then WMMA burst.
    Frag16 vf[4];
#pragma unroll
    for (int j = 0; j < 4; ++j) {
      const bf16* vb = vpT + (size_t)(16 * j + ln15) * SS + kb + kh;
      vf[j].h[0] = *(const v8bf*)(vb);
      vf[j].h[1] = *(const v8bf*)(vb + 16);
    }
#pragma unroll
    for (int j = 0; j < 4; ++j)
      acc[j] = wmma_bf16f32(pf.v, vf[j].v, acc[j]);
  }

  // softmax1 normalization and store into (B,S,D) bf16 (col = h*64 + hd).
#pragma unroll
  for (int r = 0; r < 8; ++r) {
    float denom = __expf(-m[r]) + l[r];
    int row = qrow0 + r + 8 * half;
    size_t obase = ((size_t)b * SS + row) * DD + h * HDD;
#pragma unroll
    for (int j = 0; j < 4; ++j)
      attnb[obase + 16 * j + ln15] = (bf16)(acc[j][r] / denom);
  }
}

// ---------------------------------------------------------------------------
// Residual add + LayerNorm. One block (256 thr) per token row of D=1024.
// Optional bf16 copy of the output (feeds the next GEMM without a cvt pass).
// ---------------------------------------------------------------------------
__global__ void add_ln_kernel(const float* __restrict__ a, const float* __restrict__ b,
                              const float* __restrict__ g, const float* __restrict__ be,
                              float* __restrict__ outf, bf16* __restrict__ outb) {
  __shared__ float rs[256];
  __shared__ float rq[256];
  const int row = blockIdx.x;
  const int tid = threadIdx.x;
  const float* pa = a + (size_t)row * DD;
  const float* pb = b + (size_t)row * DD;

  float s = 0.f, s2 = 0.f;
  for (int c = tid; c < DD; c += 256) {
    float v = pa[c] + pb[c];
    s += v; s2 += v * v;
  }
  rs[tid] = s; rq[tid] = s2;
  __syncthreads();
  for (int o = 128; o; o >>= 1) {
    if (tid < o) { rs[tid] += rs[tid + o]; rq[tid] += rq[tid + o]; }
    __syncthreads();
  }
  float mu  = rs[0] * (1.0f / DD);
  float var = rq[0] * (1.0f / DD) - mu * mu;
  float rstd = rsqrtf(var + 1e-5f);
  for (int c = tid; c < DD; c += 256) {
    float v = (pa[c] + pb[c] - mu) * rstd * g[c] + be[c];
    outf[(size_t)row * DD + c] = v;
    if (outb) outb[(size_t)row * DD + c] = (bf16)v;
  }
}

// ---------------------------------------------------------------------------
// Host orchestration
// ---------------------------------------------------------------------------
extern "C" void kernel_launch(void* const* d_in, const int* in_sizes, int n_in,
                              void* d_out, int out_size, void* d_ws, size_t ws_size,
                              hipStream_t stream) {
  (void)in_sizes; (void)n_in; (void)out_size; (void)ws_size;

  const float* x     = (const float*)d_in[0];
  const int*   pos   = (const int*)  d_in[1];
  const float* wq    = (const float*)d_in[2];
  const float* bq    = (const float*)d_in[3];
  const float* wk    = (const float*)d_in[4];
  const float* bk    = (const float*)d_in[5];
  const float* wv    = (const float*)d_in[6];
  const float* bv    = (const float*)d_in[7];
  const float* wo    = (const float*)d_in[8];
  const float* bo    = (const float*)d_in[9];
  const float* pbias = (const float*)d_in[10];
  const float* w1    = (const float*)d_in[11];
  const float* b1    = (const float*)d_in[12];
  const float* w2    = (const float*)d_in[13];
  const float* b2    = (const float*)d_in[14];
  const float* g1    = (const float*)d_in[15];
  const float* be1   = (const float*)d_in[16];
  const float* g2    = (const float*)d_in[17];
  const float* be2   = (const float*)d_in[18];
  float* out = (float*)d_out;

  // Workspace carve-out
  char* cur = (char*)d_ws;
  auto alloc = [&](size_t bytes) -> void* {
    void* p = (void*)cur;
    cur += (bytes + 255) & ~(size_t)255;
    return p;
  };
  const size_t nTok = (size_t)MM * DD;           // 4096*1024
  bf16* xb   = (bf16*)alloc(nTok * 2);
  bf16* wqb  = (bf16*)alloc((size_t)DD * DD * 2);
  bf16* wkb  = (bf16*)alloc((size_t)DD * DD * 2);
  bf16* wvb  = (bf16*)alloc((size_t)DD * DD * 2);
  bf16* wob  = (bf16*)alloc((size_t)DD * DD * 2);
  bf16* w1b  = (bf16*)alloc((size_t)DD * DFF_ * 2);
  bf16* w2b  = (bf16*)alloc((size_t)DFF_ * DD * 2);
  float* qf  = (float*)alloc(nTok * 4);
  float* kf  = (float*)alloc(nTok * 4);
  float* vf  = (float*)alloc(nTok * 4);
  bf16* qpb  = (bf16*)alloc((size_t)BB * HH * SS * 128 * 2);
  bf16* kpb  = (bf16*)alloc((size_t)BB * HH * SS * 128 * 2);
  bf16* vpT  = (bf16*)alloc((size_t)BB * HH * SS * 64 * 2);
  bf16* attb = (bf16*)alloc(nTok * 2);
  float* prj = (float*)alloc(nTok * 4);
  float* hres= (float*)alloc(nTok * 4);
  bf16* hb   = (bf16*)alloc(nTok * 2);
  bf16* ff1b = (bf16*)alloc((size_t)MM * DFF_ * 2);
  float* ff2 = (float*)alloc(nTok * 4);

  // 1) Convert x row-major; weights into WMMA fragment-swizzled order.
  cvt_f32_bf16 <<<4096, 256, 0, stream>>>(x,  xb,  nTok);
  cvt_swizzle_w<<<4096, 256, 0, stream>>>(wq, wqb, DD, DD);
  cvt_swizzle_w<<<4096, 256, 0, stream>>>(wk, wkb, DD, DD);
  cvt_swizzle_w<<<4096, 256, 0, stream>>>(wv, wvb, DD, DD);
  cvt_swizzle_w<<<4096, 256, 0, stream>>>(wo, wob, DD, DD);
  cvt_swizzle_w<<<8192, 256, 0, stream>>>(w1, w1b, DD, DFF_);
  cvt_swizzle_w<<<8192, 256, 0, stream>>>(w2, w2b, DFF_, DD);

  // 2) QKV projections (WMMA GEMMs); block tile 128x64.
  dim3 gD(DD / 64, MM / 128);       // (16, 32)
  gemm_bf16<<<gD, 128, 0, stream>>>(xb, wqb, bq, qf, nullptr, MM, DD, DD, 0);
  gemm_bf16<<<gD, 128, 0, stream>>>(xb, wkb, bk, kf, nullptr, MM, DD, DD, 0);
  gemm_bf16<<<gD, 128, 0, stream>>>(xb, wvb, bv, vf, nullptr, MM, DD, DD, 0);

  // 3) PoPE transform (also transposes V)
  pope_kernel<<<4096, 256, 0, stream>>>(qf, kf, vf, pos, pbias, qpb, kpb, vpT);

  // 4) Attention (WMMA flash + softmax1, causal)
  attention_pope<<<dim3(SS / 16, BB * HH), 32, 0, stream>>>(qpb, kpb, vpT, attb);

  // 5) Output projection
  gemm_bf16<<<gD, 128, 0, stream>>>(attb, wob, bo, prj, nullptr, MM, DD, DD, 0);

  // 6) Residual + LN1 (f32 kept for residual, bf16 copy feeds FFN GEMM)
  add_ln_kernel<<<MM, 256, 0, stream>>>(x, prj, g1, be1, hres, hb);

  // 7) FFN
  dim3 gF1(DFF_ / 64, MM / 128);    // (64, 32)
  gemm_bf16<<<gF1, 128, 0, stream>>>(hb, w1b, b1, nullptr, ff1b, MM, DFF_, DD, 1);
  gemm_bf16<<<gD, 128, 0, stream>>>(ff1b, w2b, b2, ff2, nullptr, MM, DD, DFF_, 0);

  // 8) Residual + LN2 -> final output (f32)
  add_ln_kernel<<<MM, 256, 0, stream>>>(hres, ff2, g2, be2, out, nullptr);
}
